// VisionModel_22462678958472
// MI455X (gfx1250) — compile-verified, matrix operand order checked by reading, not run on hardware
//
#include <hip/hip_runtime.h>
#include <hip/hip_bf16.h>

// ---------------------------------------------------------------------------
// Types for CDNA5 WMMA (wave32): v_wmma_f32_16x16x32_bf16
// ---------------------------------------------------------------------------
typedef __bf16 v16bf __attribute__((ext_vector_type(16)));
typedef float  v8f   __attribute__((ext_vector_type(8)));

__device__ __forceinline__ unsigned short f32_to_bf16(float f) {
    unsigned int u = __float_as_uint(f);
    unsigned int r = u + 0x7FFFu + ((u >> 16) & 1u);   // round-to-nearest-even
    return (unsigned short)(r >> 16);
}

__device__ __forceinline__ unsigned int pack_bf16x2(float a, float b) {
    return (unsigned int)f32_to_bf16(a) | ((unsigned int)f32_to_bf16(b) << 16);
}

__device__ __forceinline__ float gelu_exact(float x) {
    return 0.5f * x * (1.0f + erff(x * 0.7071067811865476f));
}

// ---------------------------------------------------------------------------
// Generic strided bf16-WMMA GEMM:  C[m,n] = epilogue( sum_k A[m,k]*B[k,n] )
//   A element at  A[offA + bz*bstrA + m*sAm + k*sAk]   (f32, cvt->bf16 in LDS)
//   B element at  B[offB + bz*bstrB + k*sBk + n*sBn]
//   C element at  C[offC + bz*bstrC + m*sCm + n*sCn]
// Epilogue: +bias[n] (opt), exact GELU (opt), res[m*sResM+n] + scale*acc (opt)
// Block = 256 threads = 8 waves laid out 2(M) x 4(N); wave tile 64x32.
// Staging uses float4 global loads + packed bf16 ds_store_b64 on uniform fast
// paths; bounds-checked scalar path only for edge tiles / unaligned strides.
// ---------------------------------------------------------------------------
struct GemmParams {
    const float* A;  long long offA, sAm, sAk, bstrA;
    const float* B;  long long offB, sBk, sBn, bstrB;
    float*       C;  long long offC, sCm, sCn, bstrC;
    const float* bias;            // may be null
    const float* res;             // may be null
    long long    sResM;
    const float* scale;           // may be null (scalar device ptr)
    int M, N, K;
    int act;                      // 0 = none, 1 = gelu
};

#define BM 128
#define BN 128
#define BKK 64
#define LDK 80   // padded row stride in halves (160B: 32B-aligned fragment reads)

__global__ __launch_bounds__(256) void gemm_bf16_wmma(GemmParams p) {
    __shared__ unsigned short sA[BM * LDK];
    __shared__ unsigned short sB[BN * LDK];

    const int tid  = threadIdx.x;
    const int lane = tid & 31;
    const int wave = tid >> 5;
    const int lr   = lane & 15;    // row/col within fragment
    const int lh   = lane >> 4;    // K-half selector

    const int m0 = blockIdx.y * BM;
    const int n0 = blockIdx.x * BN;
    const int wm = (wave >> 2) * 64;   // wave M offset: 0 / 64
    const int wn = (wave & 3)  * 32;   // wave N offset: 0..96

    const long long aBase = p.offA + (long long)blockIdx.z * p.bstrA;
    const long long bBase = p.offB + (long long)blockIdx.z * p.bstrB;
    const long long cBase = p.offC + (long long)blockIdx.z * p.bstrC;

    // Uniform fast-path eligibility (scalar branches, no lane divergence).
    const bool aFast  = (p.sAk == 1) && ((p.sAm & 3) == 0) && ((aBase & 3) == 0)
                      && (m0 + BM <= p.M);
    const bool bFastK = (p.sBk == 1) && ((p.sBn & 3) == 0) && ((bBase & 3) == 0)
                      && (n0 + BN <= p.N);
    const bool bFastN = (p.sBn == 1) && ((p.sBk & 3) == 0) && ((bBase & 3) == 0)
                      && (n0 + BN <= p.N);

    v8f acc[4][2];
    for (int mi = 0; mi < 4; ++mi)
        for (int ni = 0; ni < 2; ++ni)
            acc[mi][ni] = v8f{0.f,0.f,0.f,0.f,0.f,0.f,0.f,0.f};

    for (int k0 = 0; k0 < p.K; k0 += BKK) {
        const bool kFull = (k0 + BKK <= p.K);

        // ---- Stage A tile: LDS row-major [BM][BKK] ----
        if (aFast && kFull) {
            #pragma unroll
            for (int i = 0; i < 8; ++i) {              // 2048 float4 chunks / 256 thr
                int c  = tid + i * 256;
                int r  = c >> 4;                       // BKK/4 = 16 chunks per row
                int kc = (c & 15) << 2;
                float4 v = *(const float4*)(p.A + aBase
                            + (long long)(m0 + r) * p.sAm + (k0 + kc));
                uint2 u;
                u.x = pack_bf16x2(v.x, v.y);
                u.y = pack_bf16x2(v.z, v.w);
                *(uint2*)&sA[r * LDK + kc] = u;        // ds_store_b64
            }
        } else {
            #pragma unroll
            for (int i = 0; i < 32; ++i) {
                int idx = tid + i * 256;
                int r   = idx >> 6;
                int kk  = idx & 63;
                int gm  = m0 + r, gk = k0 + kk;
                float av = (gm < p.M && gk < p.K)
                         ? p.A[aBase + (long long)gm * p.sAm + (long long)gk * p.sAk]
                         : 0.f;
                sA[r * LDK + kk] = f32_to_bf16(av);
            }
        }

        // ---- Stage B tile: LDS column-major [BN][BKK] (row n holds K-run) ----
        if (bFastK && kFull) {
            #pragma unroll
            for (int i = 0; i < 8; ++i) {
                int c  = tid + i * 256;
                int r  = c >> 4;                       // n index
                int kc = (c & 15) << 2;
                float4 v = *(const float4*)(p.B + bBase
                            + (long long)(n0 + r) * p.sBn + (k0 + kc));
                uint2 u;
                u.x = pack_bf16x2(v.x, v.y);
                u.y = pack_bf16x2(v.z, v.w);
                *(uint2*)&sB[r * LDK + kc] = u;
            }
        } else if (bFastN && kFull) {
            #pragma unroll
            for (int i = 0; i < 8; ++i) {              // contiguous along n, scatter
                int c  = tid + i * 256;
                int kk = c >> 5;                       // BN/4 = 32 chunks per k-row
                int nc = (c & 31) << 2;
                float4 v = *(const float4*)(p.B + bBase
                            + (long long)(k0 + kk) * p.sBk + (n0 + nc));
                sB[(nc + 0) * LDK + kk] = f32_to_bf16(v.x);
                sB[(nc + 1) * LDK + kk] = f32_to_bf16(v.y);
                sB[(nc + 2) * LDK + kk] = f32_to_bf16(v.z);
                sB[(nc + 3) * LDK + kk] = f32_to_bf16(v.w);
            }
        } else {
            #pragma unroll
            for (int i = 0; i < 32; ++i) {
                int idx = tid + i * 256;
                int r   = idx >> 6;
                int kk  = idx & 63;
                int gn  = n0 + r, gk = k0 + kk;
                float bv = (gn < p.N && gk < p.K)
                         ? p.B[bBase + (long long)gk * p.sBk + (long long)gn * p.sBn]
                         : 0.f;
                sB[r * LDK + kk] = f32_to_bf16(bv);
            }
        }
        __syncthreads();

        #pragma unroll
        for (int ks = 0; ks < BKK; ks += 32) {
            v16bf af[4], bfr[2];
            #pragma unroll
            for (int mi = 0; mi < 4; ++mi) {
                int row = wm + mi * 16 + lr;
                af[mi] = *(const v16bf*)&sA[row * LDK + ks + lh * 16];
            }
            #pragma unroll
            for (int ni = 0; ni < 2; ++ni) {
                int col = wn + ni * 16 + lr;
                bfr[ni] = *(const v16bf*)&sB[col * LDK + ks + lh * 16];
            }
            #pragma unroll
            for (int mi = 0; mi < 4; ++mi)
                #pragma unroll
                for (int ni = 0; ni < 2; ++ni)
                    acc[mi][ni] = __builtin_amdgcn_wmma_f32_16x16x32_bf16(
                        false, af[mi], false, bfr[ni], (short)0, acc[mi][ni],
                        false, false);
        }
        __syncthreads();
    }

    const float sc = p.scale ? p.scale[0] : 1.0f;
    #pragma unroll
    for (int mi = 0; mi < 4; ++mi) {
        #pragma unroll
        for (int ni = 0; ni < 2; ++ni) {
            #pragma unroll
            for (int j = 0; j < 8; ++j) {
                int grow = m0 + wm + mi * 16 + lh * 8 + j;
                int gcol = n0 + wn + ni * 16 + lr;
                if (grow < p.M && gcol < p.N) {
                    float v = acc[mi][ni][j];
                    if (p.bias) v += p.bias[gcol];
                    if (p.act == 1) v = gelu_exact(v);
                    if (p.res) v = p.res[(long long)grow * p.sResM + gcol] + sc * v;
                    p.C[cBase + (long long)grow * p.sCm + (long long)gcol * p.sCn] = v;
                }
            }
        }
    }
}

// ---------------------------------------------------------------------------
// Direct 3x3 stride-2 pad-1 convolution (stem; <4% of total FLOPs)
// x: [B,Ci,Hi,Wi]  w: [Co,Ci,3,3]  y: [B,Co,Ho,Wo]
// ---------------------------------------------------------------------------
__global__ void conv3x3_s2(const float* __restrict__ x, const float* __restrict__ w,
                           const float* __restrict__ bias, float* __restrict__ y,
                           int Ci, int Hi, int Wi, int Co, int Ho, int Wo,
                           long long total) {
    long long idx = (long long)blockIdx.x * blockDim.x + threadIdx.x;
    if (idx >= total) return;
    int ow = (int)(idx % Wo); long long t = idx / Wo;
    int oh = (int)(t % Ho);   t /= Ho;
    int co = (int)(t % Co);   int b = (int)(t / Co);

    float acc = bias[co];
    const float* xb = x + (long long)b * Ci * Hi * Wi;
    const float* wc = w + (long long)co * Ci * 9;
    for (int ci = 0; ci < Ci; ++ci) {
        const float* xp = xb + (long long)ci * Hi * Wi;
        const float* wp = wc + ci * 9;
        #pragma unroll
        for (int kh = 0; kh < 3; ++kh) {
            int ih = oh * 2 - 1 + kh;
            if ((unsigned)ih < (unsigned)Hi) {
                #pragma unroll
                for (int kw = 0; kw < 3; ++kw) {
                    int iw = ow * 2 - 1 + kw;
                    if ((unsigned)iw < (unsigned)Wi)
                        acc += xp[(long long)ih * Wi + iw] * wp[kh * 3 + kw];
                }
            }
        }
    }
    y[idx] = acc;
}

// ---------------------------------------------------------------------------
// Stem LayerNorm: normalize over the entire (C,H,W) per image.
// ---------------------------------------------------------------------------
__global__ void stem_ln_stats(const float* __restrict__ x, long long N,
                              float* __restrict__ stats) {
    __shared__ float r1[256], r2[256];
    int tid = threadIdx.x;
    long long base = (long long)blockIdx.x * N;
    float s = 0.f, q = 0.f;
    for (long long i = tid; i < N; i += 256) {
        float v = x[base + i];
        s += v; q += v * v;
    }
    r1[tid] = s; r2[tid] = q; __syncthreads();
    for (int st = 128; st > 0; st >>= 1) {
        if (tid < st) { r1[tid] += r1[tid + st]; r2[tid] += r2[tid + st]; }
        __syncthreads();
    }
    if (tid == 0) {
        float mu  = r1[0] / (float)N;
        float var = r2[0] / (float)N - mu * mu;
        stats[2 * blockIdx.x]     = mu;
        stats[2 * blockIdx.x + 1] = rsqrtf(var + 1e-5f);
    }
}

// In-place: x = relu((x - mu)*rstd * g[h,w,c] + b[h,w,c]); x is NCHW, g/b HWC.
__global__ void stem_ln_apply_relu(float* __restrict__ x, const float* __restrict__ stats,
                                   const float* __restrict__ g, const float* __restrict__ bb,
                                   int C, int H, int Wd, long long total) {
    long long idx = (long long)blockIdx.x * blockDim.x + threadIdx.x;
    if (idx >= total) return;
    int w = (int)(idx % Wd); long long t = idx / Wd;
    int h = (int)(t % H);    t /= H;
    int c = (int)(t % C);    int b = (int)(t / C);
    float mu = stats[2 * b], rs = stats[2 * b + 1];
    long long gi = ((long long)h * Wd + w) * C + c;
    float v = (x[idx] - mu) * rs * g[gi] + bb[gi];
    x[idx] = v > 0.f ? v : 0.f;
}

// ---------------------------------------------------------------------------
// Per-token LayerNorm over D=768: one block per token.
// ---------------------------------------------------------------------------
__global__ void token_ln(const float* __restrict__ x, const float* __restrict__ g,
                         const float* __restrict__ bb, float* __restrict__ y, int Dd) {
    __shared__ float red[256];
    int tid = threadIdx.x;
    long long base = (long long)blockIdx.x * Dd;
    float s = 0.f;
    for (int i = tid; i < Dd; i += 256) s += x[base + i];
    red[tid] = s; __syncthreads();
    for (int st = 128; st > 0; st >>= 1) {
        if (tid < st) red[tid] += red[tid + st];
        __syncthreads();
    }
    float mu = red[0] / (float)Dd;
    __syncthreads();
    float vs = 0.f;
    for (int i = tid; i < Dd; i += 256) { float d = x[base + i] - mu; vs += d * d; }
    red[tid] = vs; __syncthreads();
    for (int st = 128; st > 0; st >>= 1) {
        if (tid < st) red[tid] += red[tid + st];
        __syncthreads();
    }
    float rs = rsqrtf(red[0] / (float)Dd + 1e-5f);
    __syncthreads();
    for (int i = tid; i < Dd; i += 256)
        y[base + i] = (x[base + i] - mu) * rs * g[i] + bb[i];
}

// ---------------------------------------------------------------------------
// Row softmax with 1/8 logit scale (in place). One block per row.
// ---------------------------------------------------------------------------
__global__ void softmax_row(float* __restrict__ s, int n) {
    __shared__ float red[256];
    int tid = threadIdx.x;
    long long base = (long long)blockIdx.x * n;
    float m = -3.4e38f;
    for (int i = tid; i < n; i += 256) m = fmaxf(m, s[base + i] * 0.125f);
    red[tid] = m; __syncthreads();
    for (int st = 128; st > 0; st >>= 1) {
        if (tid < st) red[tid] = fmaxf(red[tid], red[tid + st]);
        __syncthreads();
    }
    m = red[0]; __syncthreads();
    float sum = 0.f;
    for (int i = tid; i < n; i += 256) {
        float e = expf(s[base + i] * 0.125f - m);
        s[base + i] = e; sum += e;
    }
    red[tid] = sum; __syncthreads();
    for (int st = 128; st > 0; st >>= 1) {
        if (tid < st) red[tid] += red[tid + st];
        __syncthreads();
    }
    float inv = 1.f / red[0];
    __syncthreads();
    for (int i = tid; i < n; i += 256) s[base + i] *= inv;
}

// out[b,d] = mean_s x[b,s,d]
__global__ void mean_tokens(const float* __restrict__ x, float* __restrict__ out,
                            int S, int Dd, int total) {
    int idx = blockIdx.x * blockDim.x + threadIdx.x;
    if (idx >= total) return;
    int d = idx % Dd, b = idx / Dd;
    const float* p = x + (long long)b * S * Dd + d;
    float s = 0.f;
    for (int t = 0; t < S; ++t) s += p[(long long)t * Dd];
    out[idx] = s / (float)S;
}

// ---------------------------------------------------------------------------
// Host-side orchestration
// ---------------------------------------------------------------------------
static void launch_gemm(hipStream_t stream,
                        const float* A, long long offA, long long sAm, long long sAk, long long bstrA,
                        const float* B, long long offB, long long sBk, long long sBn, long long bstrB,
                        float* C, long long offC, long long sCm, long long sCn, long long bstrC,
                        const float* bias, const float* res, long long sResM, const float* scale,
                        int M, int N, int K, int act, int batches) {
    GemmParams p{A, offA, sAm, sAk, bstrA,
                 B, offB, sBk, sBn, bstrB,
                 C, offC, sCm, sCn, bstrC,
                 bias, res, sResM, scale, M, N, K, act};
    dim3 grid((N + BN - 1) / BN, (M + BM - 1) / BM, batches);
    gemm_bf16_wmma<<<grid, dim3(256), 0, stream>>>(p);
}

extern "C" void kernel_launch(void* const* d_in, const int* in_sizes, int n_in,
                              void* d_out, int out_size, void* d_ws, size_t ws_size,
                              hipStream_t stream) {
    (void)in_sizes; (void)n_in; (void)out_size; (void)ws_size;

    const int Bb = 4, S = 784, D = 768, NH = 12, HD = 64, L = 12, NTAGS = 5813;
    const int BS = Bb * S; // 3136

    // Inputs (setup_inputs insertion order; 'blocks' pytree leaves sorted by key)
    const float* image = (const float*)d_in[0];
    const float* c1w = (const float*)d_in[1];  const float* c1b = (const float*)d_in[2];
    const float* ln1g= (const float*)d_in[3];  const float* ln1b= (const float*)d_in[4];
    const float* c2w = (const float*)d_in[5];  const float* c2b = (const float*)d_in[6];
    const float* ln2g= (const float*)d_in[7];  const float* ln2b= (const float*)d_in[8];
    const float* c3w = (const float*)d_in[9];  const float* c3b = (const float*)d_in[10];
    const float* ln3g= (const float*)d_in[11]; const float* ln3b= (const float*)d_in[12];
    const float* c4w = (const float*)d_in[13]; const float* c4b = (const float*)d_in[14];
    const float* ln4g= (const float*)d_in[15]; const float* ln4b= (const float*)d_in[16];
    const float* c5w = (const float*)d_in[17]; const float* c5b = (const float*)d_in[18];
    const float* m1b_ = (const float*)d_in[19];
    const float* m1w_ = (const float*)d_in[20];
    const float* m2b_ = (const float*)d_in[21];
    const float* m2w_ = (const float*)d_in[22];
    const float* n1b_ = (const float*)d_in[23];
    const float* n1g_ = (const float*)d_in[24];
    const float* n2b_ = (const float*)d_in[25];
    const float* n2g_ = (const float*)d_in[26];
    const float* ob_  = (const float*)d_in[27];
    const float* ow_  = (const float*)d_in[28];
    const float* qb_  = (const float*)d_in[29];
    const float* qw_  = (const float*)d_in[30];
    const float* s1_  = (const float*)d_in[31];
    const float* s2_  = (const float*)d_in[32];
    const float* normg= (const float*)d_in[33];
    const float* normb= (const float*)d_in[34];
    const float* headw= (const float*)d_in[35];
    const float* headb= (const float*)d_in[36];
    float* out = (float*)d_out;

    // Workspace arena (floats), regions time-multiplexed:
    float* W    = (float*)d_ws;
    float* buf0 = W;                    // conv1 out (12,845,056) / per-b scores (7,375,872)
    float* buf1 = buf0 + 12845056;      // conv2 out / qkv (7,225,344) / mlp hidden (9,633,792)
    float* buf2 = buf1 + 9633792;       // conv3 out (3,211,264) / LN-out h / attn o (2,408,448)
    float* buf3 = buf2 + 3211264;       // conv4 out (1,605,632)
    float* tokA = buf3 + 1605632;       // tokens ping (2,408,448)
    float* tokB = tokA + 2408448;       // tokens pong
    float* stats= tokB + 2408448;       // 8 floats
    float* pooled = stats + 8;          // 3072 floats

    // ---- Stem: conv -> whole-map LN -> relu, x4 ----
    struct Stage { const float* xin; float* yout; int Ci,Hi,Co,Ho;
                   const float* cw; const float* cb; const float* g; const float* b; };
    const Stage st[4] = {
        { image, buf0,   3, 448,  64, 224, c1w, c1b, ln1g, ln1b },
        { buf0,  buf1,  64, 224, 128, 112, c2w, c2b, ln2g, ln2b },
        { buf1,  buf2, 128, 112, 256,  56, c3w, c3b, ln3g, ln3b },
        { buf2,  buf3, 256,  56, 512,  28, c4w, c4b, ln4g, ln4b },
    };
    for (int i = 0; i < 4; ++i) {
        long long total = (long long)Bb * st[i].Co * st[i].Ho * st[i].Ho;
        conv3x3_s2<<<dim3((unsigned)((total + 255) / 256)), dim3(256), 0, stream>>>(
            st[i].xin, st[i].cw, st[i].cb, st[i].yout,
            st[i].Ci, st[i].Hi, st[i].Hi, st[i].Co, st[i].Ho, st[i].Ho, total);
        long long N = (long long)st[i].Co * st[i].Ho * st[i].Ho;
        stem_ln_stats<<<dim3(Bb), dim3(256), 0, stream>>>(st[i].yout, N, stats);
        stem_ln_apply_relu<<<dim3((unsigned)((total + 255) / 256)), dim3(256), 0, stream>>>(
            st[i].yout, stats, st[i].g, st[i].b, st[i].Co, st[i].Ho, st[i].Ho, total);
    }

    // ---- conv5 (1x1, 512->768) as batched GEMM: tok[b,s,d] = sum_c x4[b,c,s]*w5[d,c]+b5
    launch_gemm(stream,
        buf3, 0, /*sAm*/1, /*sAk*/784, /*bstrA*/512LL * 784,
        c5w,  0, /*sBk*/1, /*sBn*/512, /*bstrB*/0,
        tokA, 0, /*sCm*/D, /*sCn*/1,  /*bstrC*/(long long)S * D,
        c5b, nullptr, 0, nullptr, S, D, 512, 0, Bb);

    // ---- 12 transformer blocks ----
    for (int l = 0; l < L; ++l) {
        const float* qw  = qw_ + (long long)l * D * 3 * D;
        const float* qb  = qb_ + (long long)l * 3 * D;
        const float* ow  = ow_ + (long long)l * D * D;
        const float* ob  = ob_ + (long long)l * D;
        const float* m1w = m1w_ + (long long)l * D * 4 * D;
        const float* m1b = m1b_ + (long long)l * 4 * D;
        const float* m2w = m2w_ + (long long)l * 4 * D * D;
        const float* m2b = m2b_ + (long long)l * D;

        // h = LN(x)
        token_ln<<<dim3(BS), dim3(256), 0, stream>>>(tokA, n1g_ + l * D, n1b_ + l * D, buf2, D);

        // qkv = h @ qw + qb      [BS, 2304]
        launch_gemm(stream,
            buf2, 0, D, 1, 0,
            qw,   0, 3LL * D, 1, 0,
            buf1, 0, 3LL * D, 1, 0,
            qb, nullptr, 0, nullptr, BS, 3 * D, D, 0, 1);

        // attention, per batch image (score buffer = 12*784*784 f32)
        for (int b = 0; b < Bb; ++b) {
            const float* qkvb = buf1 + (long long)b * S * 3 * D;
            // scores[h,s,t] = sum_d q[s,h,d] * k[t,h,d]     (grid.z = heads)
            launch_gemm(stream,
                qkvb, 0,       3LL * D, 1,       HD,
                qkvb, D,       1,       3LL * D, HD,
                buf0, 0,       S,       1,       (long long)S * S,
                nullptr, nullptr, 0, nullptr, S, S, HD, 0, NH);
            softmax_row<<<dim3(NH * S), dim3(256), 0, stream>>>(buf0, S);
            // o[s,h,d] = sum_t P[h,s,t] * v[t,h,d]
            launch_gemm(stream,
                buf0, 0,            S,       1,  (long long)S * S,
                qkvb, 2LL * D,      3LL * D, 1,  HD,
                buf2, (long long)b * S * D, D, 1, HD,
                nullptr, nullptr, 0, nullptr, S, HD, S, 0, NH);
        }

        // x = x + (o @ ow + ob) * s1   -> tokB
        launch_gemm(stream,
            buf2, 0, D, 1, 0,
            ow,   0, D, 1, 0,
            tokB, 0, D, 1, 0,
            ob, tokA, D, s1_ + l, BS, D, D, 0, 1);

        // h = LN(x)
        token_ln<<<dim3(BS), dim3(256), 0, stream>>>(tokB, n2g_ + l * D, n2b_ + l * D, buf2, D);

        // hid = gelu(h @ m1w + m1b)    [BS, 3072]
        launch_gemm(stream,
            buf2, 0, D, 1, 0,
            m1w,  0, 4LL * D, 1, 0,
            buf1, 0, 4LL * D, 1, 0,
            m1b, nullptr, 0, nullptr, BS, 4 * D, D, 1, 1);

        // x = x + (hid @ m2w + m2b) * s2   -> tokA
        launch_gemm(stream,
            buf1, 0, 4LL * D, 1, 0,
            m2w,  0, D, 1, 0,
            tokA, 0, D, 1, 0,
            m2b, tokB, D, s2_ + l, BS, D, 4 * D, 0, 1);
    }

    // ---- final LN, mean pool, head ----
    token_ln<<<dim3(BS), dim3(256), 0, stream>>>(tokA, normg, normb, buf2, D);
    mean_tokens<<<dim3((Bb * D + 255) / 256), dim3(256), 0, stream>>>(buf2, pooled, S, D, Bb * D);
    launch_gemm(stream,
        pooled, 0, D, 1, 0,
        headw,  0, NTAGS, 1, 0,
        out,    0, NTAGS, 1, 0,
        headb, nullptr, 0, nullptr, Bb, NTAGS, D, 0, 1);
}
